// MLP_26164940767965
// MI455X (gfx1250) — matrix operand order repaired and checked by reading, not compiled, not run on hardware
//
#include <hip/hip_runtime.h>
#include <math.h>

typedef unsigned short u16;
typedef unsigned int   u32;

typedef __bf16 v16bf __attribute__((ext_vector_type(16)));
typedef float  v8f   __attribute__((ext_vector_type(8)));
typedef float  v4f   __attribute__((ext_vector_type(4)));
typedef u32    v4u   __attribute__((ext_vector_type(4)));
typedef u32    v2u   __attribute__((ext_vector_type(2)));

#define E_MAXC 768
#define H_MAXC 3072
#define NTOK   16384   // 8 * 2048 tokens

union Frag {
    v16bf v;
    v4u   q[2];
};

__device__ __forceinline__ u16 bf16_rne(float f) {
    u32 u = __float_as_uint(f);
    u32 r = u + 0x7FFFu + ((u >> 16) & 1u);   // round-to-nearest-even
    return (u16)(r >> 16);
}

__device__ __forceinline__ void bf16_split(float f, u16& hi, u16& lo) {
    hi = bf16_rne(f);
    float fh = __uint_as_float(((u32)hi) << 16);
    lo = bf16_rne(f - fh);
}

// tanh-GELU rewritten branch-free: 0.5x(1+tanh(y)) == x * sigmoid(2y)
__device__ __forceinline__ float gelu_tanh(float x) {
    const float c = 0.7978845608028654f;      // sqrt(2/pi)
    float y = c * (x + 0.044715f * x * x * x);
    return x / (1.0f + __expf(-2.0f * y));
}

// Analytic alpha-mixed coefficient C0[h,e] (== C1[e,h]); also returns the
// bias coefficient c0b[h] (h-mask only) through cb0.
__device__ __forceinline__ float mix_coef(int h, int e,
                                          const float* __restrict__ alpha_e,
                                          const float* __restrict__ alpha_r,
                                          float& cb0) {
    const int   Ech[3] = {384, 576, 768};
    const int   Rch[3] = {2, 3, 4};
    float coef = 0.f;
    cb0 = 0.f;
#pragma unroll
    for (int i = 0; i < 3; ++i) {
        float s = 0.f;
#pragma unroll
        for (int j = 0; j < 3; ++j)
            if (h < Rch[j] * Ech[i]) s += alpha_r[j];
        float t = alpha_e[i] * s;
        cb0 += t;
        if (e < Ech[i]) coef += t;
    }
    return coef;
}

// ---------------------------------------------------------------------------
// Prep kernels: fold masks into weights (coalesced in each weight's layout),
// emitting bf16 hi/lo pairs for the bf16x3 WMMA scheme.
// ---------------------------------------------------------------------------
__global__ __launch_bounds__(256) void prep_w0_kernel(
    const float* __restrict__ w_fc, const float* __restrict__ b_fc,
    const float* __restrict__ alpha_e, const float* __restrict__ alpha_r,
    u16* __restrict__ W0h, u16* __restrict__ W0l, float* __restrict__ b0)
{
    int idx = blockIdx.x * 256 + threadIdx.x;      // idx = h*E + e (e fast)
    if (idx >= H_MAXC * E_MAXC) return;
    int h = idx / E_MAXC;
    int e = idx - h * E_MAXC;
    float cb0;
    float coef = mix_coef(h, e, alpha_e, alpha_r, cb0);
    u16 hi, lo;
    bf16_split(w_fc[idx] * coef, hi, lo);
    W0h[idx] = hi; W0l[idx] = lo;
    if (e == 0) b0[h] = b_fc[h] * cb0;
}

__global__ __launch_bounds__(256) void prep_w1_kernel(
    const float* __restrict__ w_proj, const float* __restrict__ b_proj,
    const float* __restrict__ alpha_e, const float* __restrict__ alpha_r,
    u16* __restrict__ W1h, u16* __restrict__ W1l, float* __restrict__ b1)
{
    int idx = blockIdx.x * 256 + threadIdx.x;      // idx = e*H + h (h fast)
    if (idx >= E_MAXC * H_MAXC) return;
    int e = idx / H_MAXC;
    int h = idx - e * H_MAXC;
    float cb0;
    float coef = mix_coef(h, e, alpha_e, alpha_r, cb0);
    u16 hi, lo;
    bf16_split(w_proj[idx] * coef, hi, lo);
    W1h[idx] = hi; W1l[idx] = lo;
    if (h == 0) {
        const int Ech[3] = {384, 576, 768};
        float cb1 = 0.f;
#pragma unroll
        for (int i = 0; i < 3; ++i) if (e < Ech[i]) cb1 += alpha_e[i];
        cb1 *= (alpha_r[0] + alpha_r[1] + alpha_r[2]);
        b1[e] = b_proj[e] * cb1;
    }
}

// ---------------------------------------------------------------------------
// One-time f32 -> bf16 hi/lo split of x (done once, not once per N-block).
// ---------------------------------------------------------------------------
__global__ __launch_bounds__(256) void splitx_kernel(
    const float* __restrict__ x, u16* __restrict__ Xhi, u16* __restrict__ Xlo)
{
    int idx = (blockIdx.x * 256 + threadIdx.x) * 4;
    if (idx >= NTOK * E_MAXC) return;
    v4f f = *(const v4f*)(x + idx);
    u16 h[4], l[4];
#pragma unroll
    for (int i = 0; i < 4; ++i) bf16_split(f[i], h[i], l[i]);
    v2u ph = { (u32)h[0] | ((u32)h[1] << 16), (u32)h[2] | ((u32)h[3] << 16) };
    v2u pl = { (u32)l[0] | ((u32)l[1] << 16), (u32)l[2] | ((u32)l[3] << 16) };
    *(v2u*)(Xhi + idx) = ph;
    *(v2u*)(Xlo + idx) = pl;
}

// ---------------------------------------------------------------------------
// Tiled GEMM, both operands K-major (A: [M,K], B: [N,K]), bf16x3 precision
// via v_wmma_f32_16x16x32_bf16. 256 threads = 8 waves; block tile 128x128,
// K-step 32; wave grid 4(M) x 2(N), per-wave 2x4 of 16x16 accumulators.
// Ping-pong LDS double buffering: global prefetch to regs issued before the
// single per-iteration barrier, stores go to the other buffer.
// MODE 0: epilogue = bias + GELU -> h hi/lo.   MODE 1: bias -> f32 out.
// ---------------------------------------------------------------------------
template<int MODE>
__global__ __launch_bounds__(256) void gemm_kernel(
    const u16* __restrict__ Ahi, const u16* __restrict__ Alo,
    const u16* __restrict__ Bhi, const u16* __restrict__ Blo,
    const float* __restrict__ bias,
    u16* __restrict__ OutHi, u16* __restrict__ OutLo,
    float* __restrict__ OutF)
{
    constexpr int KDIM   = (MODE == 0) ? E_MAXC : H_MAXC;
    constexpr int OSTR   = (MODE == 0) ? H_MAXC : E_MAXC;
    constexpr int KSTEPS = KDIM / 32;

    __shared__ u16 sAh[2][128 * 32];
    __shared__ u16 sAl[2][128 * 32];
    __shared__ u16 sBh[2][128 * 32];
    __shared__ u16 sBl[2][128 * 32];

    const int tid  = threadIdx.x;
    const int nb   = blockIdx.x;
    const int mb   = blockIdx.y;
    const int lane = tid & 31;
    const int wave = tid >> 5;
    const int wm   = wave & 3;     // 4 M-groups of 32 rows
    const int wn   = wave >> 2;    // 2 N-groups of 64 cols
    const int lr   = lane & 15;
    const int half = lane >> 4;
    const int kofA = half * 8;     // A frag: K = 16*(e>=8) + 8*half + (e&7)
    const int kofB = half * 8;     // B frag: K = 16*(e>=8) + 8*half + (e&7)

    const int srow = tid >> 1;           // staging row 0..127
    const int scg  = (tid & 1) * 16;     // staging K sub-group 0 or 16
    const int sdst = srow * 32 + scg;

    const u16* gAh = Ahi + (size_t)(mb * 128 + srow) * KDIM + scg;
    const u16* gAl = Alo + (size_t)(mb * 128 + srow) * KDIM + scg;
    const u16* gBh = Bhi + (size_t)(nb * 128 + srow) * KDIM + scg;
    const u16* gBl = Blo + (size_t)(nb * 128 + srow) * KDIM + scg;

    v8f acc[2][4];
#pragma unroll
    for (int mt = 0; mt < 2; ++mt)
#pragma unroll
        for (int nt = 0; nt < 4; ++nt)
#pragma unroll
            for (int r = 0; r < 8; ++r) acc[mt][nt][r] = 0.f;

    v4u pf[8];
    auto fetch = [&](int kt) {
        const int o = kt * 32;
        pf[0] = *(const v4u*)(gAh + o);  pf[1] = *(const v4u*)(gAh + o + 8);
        pf[2] = *(const v4u*)(gAl + o);  pf[3] = *(const v4u*)(gAl + o + 8);
        pf[4] = *(const v4u*)(gBh + o);  pf[5] = *(const v4u*)(gBh + o + 8);
        pf[6] = *(const v4u*)(gBl + o);  pf[7] = *(const v4u*)(gBl + o + 8);
    };
    auto stash = [&](int buf) {
        *(v4u*)&sAh[buf][sdst] = pf[0];  *(v4u*)&sAh[buf][sdst + 8] = pf[1];
        *(v4u*)&sAl[buf][sdst] = pf[2];  *(v4u*)&sAl[buf][sdst + 8] = pf[3];
        *(v4u*)&sBh[buf][sdst] = pf[4];  *(v4u*)&sBh[buf][sdst + 8] = pf[5];
        *(v4u*)&sBl[buf][sdst] = pf[6];  *(v4u*)&sBl[buf][sdst + 8] = pf[7];
    };

    fetch(0);
    stash(0);

    for (int kt = 0; kt < KSTEPS; ++kt) {
        const int cur = kt & 1;
        if (kt + 1 < KSTEPS) fetch(kt + 1);   // overlap global latency w/ WMMA
        __syncthreads();                       // sX[cur] visible to all waves

        Frag fAh[2], fAl[2], fBh[4], fBl[4];
#pragma unroll
        for (int mt = 0; mt < 2; ++mt) {
            int base = (wm * 32 + mt * 16 + lr) * 32;
            fAh[mt].q[0] = *(const v4u*)&sAh[cur][base + kofA];
            fAh[mt].q[1] = *(const v4u*)&sAh[cur][base + kofA + 16];
            fAl[mt].q[0] = *(const v4u*)&sAl[cur][base + kofA];
            fAl[mt].q[1] = *(const v4u*)&sAl[cur][base + kofA + 16];
        }
#pragma unroll
        for (int nt = 0; nt < 4; ++nt) {
            int base = (wn * 64 + nt * 16 + lr) * 32;
            fBh[nt].q[0] = *(const v4u*)&sBh[cur][base + kofB];
            fBh[nt].q[1] = *(const v4u*)&sBh[cur][base + kofB + 16];
            fBl[nt].q[0] = *(const v4u*)&sBl[cur][base + kofB];
            fBl[nt].q[1] = *(const v4u*)&sBl[cur][base + kofB + 16];
        }

        // bf16x3: hi*hi + hi*lo + lo*hi  (24 WMMAs per K-step per wave)
#pragma unroll
        for (int mt = 0; mt < 2; ++mt)
#pragma unroll
            for (int nt = 0; nt < 4; ++nt) {
                acc[mt][nt] = __builtin_amdgcn_wmma_f32_16x16x32_bf16(
                    false, fAh[mt].v, false, fBh[nt].v, (short)0, acc[mt][nt], false, false);
                acc[mt][nt] = __builtin_amdgcn_wmma_f32_16x16x32_bf16(
                    false, fAh[mt].v, false, fBl[nt].v, (short)0, acc[mt][nt], false, false);
                acc[mt][nt] = __builtin_amdgcn_wmma_f32_16x16x32_bf16(
                    false, fAl[mt].v, false, fBh[nt].v, (short)0, acc[mt][nt], false, false);
            }

        if (kt + 1 < KSTEPS) stash(cur ^ 1);  // other buffer: no extra barrier
    }

    // epilogue (C/D layout: lane N = lane&15, VGPR r -> M = r + 8*(lane>=16))
#pragma unroll
    for (int nt = 0; nt < 4; ++nt) {
        int n = nb * 128 + wn * 64 + nt * 16 + lr;
        float bv = bias[n];
#pragma unroll
        for (int mt = 0; mt < 2; ++mt) {
#pragma unroll
            for (int r = 0; r < 8; ++r) {
                int m = mb * 128 + wm * 32 + mt * 16 + half * 8 + r;
                float v = acc[mt][nt][r] + bv;
                if constexpr (MODE == 0) {
                    v = gelu_tanh(v);
                    u16 hi, lo; bf16_split(v, hi, lo);
                    OutHi[(size_t)m * OSTR + n] = hi;
                    OutLo[(size_t)m * OSTR + n] = lo;
                } else {
                    OutF[(size_t)m * OSTR + n] = v;
                }
            }
        }
    }
}

// ---------------------------------------------------------------------------
extern "C" void kernel_launch(void* const* d_in, const int* in_sizes, int n_in,
                              void* d_out, int out_size, void* d_ws, size_t ws_size,
                              hipStream_t stream)
{
    (void)in_sizes; (void)n_in; (void)out_size; (void)ws_size;
    const float* x      = (const float*)d_in[0];
    const float* w_fc   = (const float*)d_in[1];
    const float* b_fc   = (const float*)d_in[2];
    const float* w_proj = (const float*)d_in[3];
    const float* b_proj = (const float*)d_in[4];
    const float* a_e    = (const float*)d_in[5];
    const float* a_r    = (const float*)d_in[6];

    char* ws = (char*)d_ws;
    size_t off = 0;
    auto take = [&](size_t bytes) -> void* {
        void* p = ws + off;
        off = (off + bytes + 255) & ~(size_t)255;
        return p;
    };
    u16*   W0h = (u16*)take((size_t)H_MAXC * E_MAXC * 2);
    u16*   W0l = (u16*)take((size_t)H_MAXC * E_MAXC * 2);
    u16*   W1h = (u16*)take((size_t)E_MAXC * H_MAXC * 2);
    u16*   W1l = (u16*)take((size_t)E_MAXC * H_MAXC * 2);
    float* b0  = (float*)take((size_t)H_MAXC * 4);
    float* b1  = (float*)take((size_t)E_MAXC * 4);
    u16*   Xhi = (u16*)take((size_t)NTOK * E_MAXC * 2);
    u16*   Xlo = (u16*)take((size_t)NTOK * E_MAXC * 2);
    u16*   Hhi = (u16*)take((size_t)NTOK * H_MAXC * 2);
    u16*   Hlo = (u16*)take((size_t)NTOK * H_MAXC * 2);

    int wBlocks = (H_MAXC * E_MAXC + 255) / 256;
    prep_w0_kernel<<<wBlocks, 256, 0, stream>>>(w_fc, b_fc, a_e, a_r, W0h, W0l, b0);
    prep_w1_kernel<<<wBlocks, 256, 0, stream>>>(w_proj, b_proj, a_e, a_r, W1h, W1l, b1);

    int xBlocks = (NTOK * E_MAXC / 4 + 255) / 256;
    splitx_kernel<<<xBlocks, 256, 0, stream>>>(x, Xhi, Xlo);

    dim3 g1(H_MAXC / 128, NTOK / 128);   // (24, 128)
    gemm_kernel<0><<<g1, 256, 0, stream>>>(Xhi, Xlo, W0h, W0l, b0,
                                           Hhi, Hlo, nullptr);

    dim3 g2(E_MAXC / 128, NTOK / 128);   // (6, 128)
    gemm_kernel<1><<<g2, 256, 0, stream>>>(Hhi, Hlo, W1h, W1l, b1,
                                           nullptr, nullptr, (float*)d_out);
}
